// NonCausalSelfAttention_29927332118932
// MI455X (gfx1250) — matrix operand order repaired and checked by reading, compiled, not run on hardware
//
#include <hip/hip_runtime.h>

// Problem dims (fixed by the reference)
#define Bdim 4
#define Tdim 2048
#define Cdim 1024
#define Hdim 16
#define Ddim 64

typedef __attribute__((ext_vector_type(16))) __bf16 v16bf;
typedef __attribute__((ext_vector_type(8)))  float  v8f;

// ---- TDM availability (device pass only; arity differs per toolchain) ----
#if defined(__HIP_DEVICE_COMPILE__) && \
    __has_builtin(__builtin_amdgcn_tensor_load_to_lds) && \
    __has_builtin(__builtin_amdgcn_s_wait_tensorcnt)
#define HAVE_TDM 1
#else
#define HAVE_TDM 0
#endif

#if __has_include(<hip/amd_detail/amd_gfx1250_TDM.h>)
#define TDM_ARITY6 1
#else
#define TDM_ARITY6 0
#endif

// Issue one TDM 2-D tile load: tileW elements (bf16) per row, tileH rows,
// global row stride = rowStride elements, LDS rows padded by 1 dword every
// (1<<padCode)*2 dwords stored (padCode: 3 -> every 16 dwords, 4 -> every 32).
__device__ __forceinline__ void tdm_load_2d(const void* gptr, unsigned int ldsOff,
                                            int tileW, int tileH, int rowStride,
                                            int padCode) {
#if HAVE_TDM
  typedef __attribute__((ext_vector_type(4))) unsigned int u32x4;
  typedef __attribute__((ext_vector_type(8))) int i32x8;
  typedef __attribute__((ext_vector_type(4))) int i32x4;
  const unsigned long long ga = (unsigned long long)(uintptr_t)gptr;
  // D# group 0: count=1 | lds_addr | global_addr[56:0] | type=2
  u32x4 g0 = { 1u, ldsOff, (unsigned int)ga,
               (unsigned int)((ga >> 32) & 0x01FFFFFFull) | 0x80000000u };
  // D# group 1: data_size=2B, pad_enable, pad_interval=padCode, pad_amount=1dw;
  // tensor_dim0=tileW, tensor_dim1=tileH, tile_dim0=tileW, tile_dim1=tileH,
  // tensor_dim0_stride=rowStride (elements)
  i32x8 g1 = { (int)((1u << 16) | (1u << 20) | ((unsigned)padCode << 22)),
               (int)(((unsigned)tileW & 0xFFFFu) << 16),
               (int)(((unsigned)tileH & 0xFFFFu) << 16),
               (int)(((unsigned)tileW & 0xFFFFu) << 16),
               (int)((unsigned)tileH & 0xFFFFu),
               rowStride, 0, 0 };
  i32x4 gz = {0, 0, 0, 0};
#if TDM_ARITY6
  i32x8 gz8 = {0, 0, 0, 0, 0, 0, 0, 0};
  __builtin_amdgcn_tensor_load_to_lds(g0, g1, gz, gz, gz8, 0);
#else
  __builtin_amdgcn_tensor_load_to_lds(g0, g1, gz, gz, 0);
#endif
#else
  (void)gptr; (void)ldsOff; (void)tileW; (void)tileH; (void)rowStride; (void)padCode;
#endif
}

__device__ __forceinline__ unsigned int lds_offset_of(const void* p) {
  // AMDGPU flat addresses map to LDS by their low 32 bits (ISA §10.2).
  return (unsigned int)(uintptr_t)p;
}

__device__ __forceinline__ unsigned short f2bf(float f) {
  unsigned int u = __float_as_uint(f);
  u += 0x7FFFu + ((u >> 16) & 1u);   // round-to-nearest-even
  return (unsigned short)(u >> 16);
}

__device__ __forceinline__ v8f wmma_bf16(v16bf a, v16bf b, v8f c) {
  // D(f32, 16x16) = A(bf16, 16x32) * B(bf16, 32x16) + C
  return __builtin_amdgcn_wmma_f32_16x16x32_bf16(false, a, false, b, (short)0, c, false, false);
}

// A-fragment (16x32 bf16, MxK) from an LDS tile stored row-major (strideDW dwords/row).
__device__ __forceinline__ v16bf load_frag_a(const unsigned int* base, int strideDW,
                                             int kOffDW, int lane) {
  const int m = lane & 15, hf = lane >> 4;
  union { v16bf v; unsigned int u[8]; } f;
  const unsigned int* row = base + m * strideDW + kOffDW;
#pragma unroll
  for (int j = 0; j < 8; ++j)
    f.u[j] = row[((j >> 2) << 3) + (hf << 2) + (j & 3)];
  return f.v;
}

// B-fragment (32x16 bf16, KxN) from an LDS tile stored N-major: Bt[n][k].
__device__ __forceinline__ v16bf load_frag_b(const unsigned int* base, int strideDW,
                                             int kOffDW, int lane) {
  const int n = lane & 15, hf = lane >> 4;
  union { v16bf v; unsigned int u[8]; } f;
  const unsigned int* row = base + n * strideDW + kOffDW;
#pragma unroll
  for (int j = 0; j < 8; ++j) f.u[j] = row[((j >> 2) << 3) + (hf << 2) + (j & 3)];
  return f.v;
}

__global__ void cvt_f32_bf16(const float* __restrict__ in, unsigned short* __restrict__ out, int n) {
  int i = blockIdx.x * blockDim.x + threadIdx.x;
  int stride = gridDim.x * blockDim.x;
  for (; i < n; i += stride) out[i] = f2bf(in[i]);
}

// ----------------------------------------------------------------------------
// GEMM: C[M,N] = A[M,K](bf16) * B[K,N](bf16) + bias
// Block tile 128x128, K-step 32. 256 threads = 8 wave32s, each a 32x64 sub-tile.
// A tiles: TDM double-buffered into padded LDS rows; B tiles: manual transpose.
// MODE 0: QKV epilogue (scatter bf16 into q/k/v [B,H,T,D], q pre-scaled by 1/8)
// MODE 1: projection epilogue (f32 out)
// ----------------------------------------------------------------------------
#define BM 128
#define BN 128
#define BK 32
#define LDA_DW 17   // 32 bf16 = 16 dwords, +1 pad (TDM pad_interval code 3)
#define LDB_DW 17

template <int MODE>
__global__ __launch_bounds__(256) void gemm_bf16(
    const unsigned short* __restrict__ A,
    const unsigned short* __restrict__ Bw,
    const float* __restrict__ bias,
    unsigned short* __restrict__ qO,
    unsigned short* __restrict__ kO,
    unsigned short* __restrict__ vO,
    float* __restrict__ outF,
    int N, int K)
{
  __shared__ unsigned int sA[2][BM * LDA_DW];
  __shared__ unsigned int sB[BN * LDB_DW];   // transposed: Bt[n][k]
  const int tid = threadIdx.x;
  const int lane = tid & 31, wave = tid >> 5;
  const int waveM = wave >> 1, waveN = wave & 1;
  const int m0 = blockIdx.y * BM, n0 = blockIdx.x * BN;

  const v8f vzero = {0.f, 0.f, 0.f, 0.f, 0.f, 0.f, 0.f, 0.f};
  v8f acc[2][4];
#pragma unroll
  for (int i = 0; i < 2; ++i)
#pragma unroll
    for (int j = 0; j < 4; ++j) acc[i][j] = vzero;

  const int bRow = tid >> 3, bCol = (tid & 7) << 4;    // B: 32 k-rows x (8x16 bf16)
  unsigned short* sBt = (unsigned short*)sB;
  const int kTiles = K / BK;

#if HAVE_TDM
  if (wave == 0)
    tdm_load_2d(A + (size_t)m0 * K, lds_offset_of(&sA[0][0]), BK, BM, K, 3);
#endif

  for (int kt = 0; kt < kTiles; ++kt) {
    const int k0 = kt * BK;
    unsigned int* sAcur = sA[kt & 1];
#if !HAVE_TDM
    {   // manual A tile staging (128x32 bf16, row-major)
      const int aRow = tid >> 1, aHalf = tid & 1;
      const uint4* g = (const uint4*)(A + (size_t)(m0 + aRow) * K + k0 + aHalf * 16);
      uint4 t0 = g[0], t1 = g[1];
      unsigned int* dst = sAcur + aRow * LDA_DW + aHalf * 8;
      dst[0]=t0.x; dst[1]=t0.y; dst[2]=t0.z; dst[3]=t0.w;
      dst[4]=t1.x; dst[5]=t1.y; dst[6]=t1.z; dst[7]=t1.w;
    }
#endif
    {   // B tile (32x128 bf16) transposed into Bt[n][k]
      const uint4* g = (const uint4*)(Bw + (size_t)(k0 + bRow) * N + n0 + bCol);
      uint4 t0 = g[0], t1 = g[1];
      unsigned int w[8] = {t0.x, t0.y, t0.z, t0.w, t1.x, t1.y, t1.z, t1.w};
#pragma unroll
      for (int j = 0; j < 8; ++j) {
        sBt[(bCol + 2*j    ) * (2*LDB_DW) + bRow] = (unsigned short)(w[j] & 0xFFFFu);
        sBt[(bCol + 2*j + 1) * (2*LDB_DW) + bRow] = (unsigned short)(w[j] >> 16);
      }
      if (kt + 1 < kTiles)
        __builtin_prefetch((const void*)(Bw + (size_t)(k0 + BK + bRow) * N + n0 + bCol), 0, 3);
    }
#if HAVE_TDM
    if (wave == 0) {
      if (kt + 1 < kTiles) {
        tdm_load_2d(A + (size_t)m0 * K + (k0 + BK),
                    lds_offset_of(&sA[(kt + 1) & 1][0]), BK, BM, K, 3);
        __builtin_amdgcn_s_wait_tensorcnt(1);   // tile kt complete (in-order)
      } else {
        __builtin_amdgcn_s_wait_tensorcnt(0);
      }
    }
#endif
    __syncthreads();

    const unsigned int* aBase = sAcur + (waveM * 32) * LDA_DW;
    const unsigned int* bBase = sB + (waveN * 64) * LDB_DW;
    v16bf a0 = load_frag_a(aBase, LDA_DW, 0, lane);
    v16bf a1 = load_frag_a(aBase + 16 * LDA_DW, LDA_DW, 0, lane);
#pragma unroll
    for (int nf = 0; nf < 4; ++nf) {
      v16bf bfr = load_frag_b(bBase + nf * 16 * LDB_DW, LDB_DW, 0, lane);
      acc[0][nf] = wmma_bf16(a0, bfr, acc[0][nf]);
      acc[1][nf] = wmma_bf16(a1, bfr, acc[1][nf]);
    }
    __syncthreads();
  }

  // ---- epilogue ----  C layout: vgpr r, lane -> (m = r + (lane>>4)*8, n = lane&15)
  const int hf = lane >> 4, ln = lane & 15;
#pragma unroll
  for (int mf = 0; mf < 2; ++mf)
#pragma unroll
    for (int nf = 0; nf < 4; ++nf)
#pragma unroll
      for (int r = 0; r < 8; ++r) {
        const int m = m0 + waveM * 32 + mf * 16 + r + hf * 8;
        const int n = n0 + waveN * 64 + nf * 16 + ln;
        float val = acc[mf][nf][r] + bias[n];
        if (MODE == 0) {
          const int sec = n >> 10;            // /C : 0=q 1=k 2=v
          const int c = n & (Cdim - 1);
          const int h = c >> 6, d = c & (Ddim - 1);
          const int b = m >> 11, t = m & (Tdim - 1);
          const size_t idx = ((size_t)(b * Hdim + h) * Tdim + t) * Ddim + d;
          unsigned short bv = f2bf(sec == 0 ? val * 0.125f : val); // fold 1/sqrt(D) into q
          if (sec == 0)      qO[idx] = bv;
          else if (sec == 1) kO[idx] = bv;
          else               vO[idx] = bv;
        } else {
          outF[(size_t)m * N + n] = val;
        }
      }
}

// ----------------------------------------------------------------------------
// Flash attention: grid (T/64, H, B), 128 threads = 4 waves, 16 q-rows/wave.
// K tiles: TDM double-buffered (row-major [key][d] IS the Bt layout for Q*K^T,
// +1dw row padding via pad_interval code 4). V staged transposed ([d][key]) by
// VALU. Online softmax across the 16 lanes of a half-wave.
// ----------------------------------------------------------------------------
#define AT_LDK 33   // 64 bf16 = 32 dwords, +1 pad
#define NKT (Tdim / 64)

__global__ __launch_bounds__(128) void attn_fa(
    const unsigned short* __restrict__ Q,
    const unsigned short* __restrict__ Kg,
    const unsigned short* __restrict__ Vg,
    unsigned short* __restrict__ Y)
{
  __shared__ unsigned int sK[2][64 * AT_LDK];
  __shared__ unsigned int sV[64 * AT_LDK];       // transposed: [d][key]
  __shared__ unsigned int sP[4][16 * AT_LDK];    // per-wave P tile [m][key]

  const int tid = threadIdx.x, lane = tid & 31, wave = tid >> 5;
  const int b = blockIdx.z, h = blockIdx.y, qt = blockIdx.x;
  const size_t headOff = (size_t)(b * Hdim + h) * Tdim * Ddim;
  const unsigned short* Qh = Q + headOff;
  const unsigned short* Kp = Kg + headOff;
  const unsigned short* Vp = Vg + headOff;
  const int qRow0 = qt * 64 + wave * 16;
  const int hf = lane >> 4, ln = lane & 15;

  // Q fragments (A layout) loaded once, directly from global (row = 32 dwords)
  v16bf qfrag0, qfrag1;
  {
    union { v16bf v; unsigned int u[8]; } f0, f1;
    const unsigned int* qrow = (const unsigned int*)(Qh + (size_t)(qRow0 + ln) * Ddim);
#pragma unroll
    for (int j = 0; j < 8; ++j) {
      const int kdw = ((j >> 2) << 3) + (hf << 2) + (j & 3);
      f0.u[j] = qrow[kdw];
      f1.u[j] = qrow[16 + kdw];
    }
    qfrag0 = f0.v; qfrag1 = f1.v;
  }

  const v8f vzero = {0.f, 0.f, 0.f, 0.f, 0.f, 0.f, 0.f, 0.f};
  float mrun[8], lrun[8];
  v8f oacc[4];
#pragma unroll
  for (int r = 0; r < 8; ++r) { mrun[r] = -1e30f; lrun[r] = 0.f; }
#pragma unroll
  for (int d = 0; d < 4; ++d) oacc[d] = vzero;

  const int ldRow = tid >> 1, ldHalf = tid & 1;  // 64 rows x (2 x 32 bf16)
  unsigned short* sVt = (unsigned short*)sV;
  unsigned short* pW  = (unsigned short*)sP[wave];

#if HAVE_TDM
  if (wave == 0)
    tdm_load_2d(Kp, lds_offset_of(&sK[0][0]), Ddim, 64, Ddim, 4);
#endif

  for (int ktile = 0; ktile < NKT; ++ktile) {
    const int key0 = ktile * 64;
    unsigned int* sKt = sK[ktile & 1];
#if !HAVE_TDM
    {   // manual K tile staging: row-major [key][d]
      const uint4* g = (const uint4*)(Kp + (size_t)(key0 + ldRow) * Ddim + ldHalf * 32);
      uint4 t0 = g[0], t1 = g[1], t2 = g[2], t3 = g[3];
      unsigned int* dst = sKt + ldRow * AT_LDK + ldHalf * 16;
      dst[0]=t0.x;  dst[1]=t0.y;  dst[2]=t0.z;  dst[3]=t0.w;
      dst[4]=t1.x;  dst[5]=t1.y;  dst[6]=t1.z;  dst[7]=t1.w;
      dst[8]=t2.x;  dst[9]=t2.y;  dst[10]=t2.z; dst[11]=t2.w;
      dst[12]=t3.x; dst[13]=t3.y; dst[14]=t3.z; dst[15]=t3.w;
    }
#endif
    {   // V tile transposed: [d][key]
      const uint4* g = (const uint4*)(Vp + (size_t)(key0 + ldRow) * Ddim + ldHalf * 32);
      uint4 t0 = g[0], t1 = g[1], t2 = g[2], t3 = g[3];
      unsigned int w[16] = {t0.x,t0.y,t0.z,t0.w, t1.x,t1.y,t1.z,t1.w,
                            t2.x,t2.y,t2.z,t2.w, t3.x,t3.y,t3.z,t3.w};
#pragma unroll
      for (int j = 0; j < 16; ++j) {
        const int d0 = ldHalf * 32 + 2 * j;
        sVt[(d0    ) * (2*AT_LDK) + ldRow] = (unsigned short)(w[j] & 0xFFFFu);
        sVt[(d0 + 1) * (2*AT_LDK) + ldRow] = (unsigned short)(w[j] >> 16);
      }
    }
#if HAVE_TDM
    if (wave == 0) {
      if (ktile + 1 < NKT) {
        tdm_load_2d(Kp + (size_t)(key0 + 64) * Ddim,
                    lds_offset_of(&sK[(ktile + 1) & 1][0]), Ddim, 64, Ddim, 4);
        __builtin_amdgcn_s_wait_tensorcnt(1);   // tile ktile complete (in-order)
      } else {
        __builtin_amdgcn_s_wait_tensorcnt(0);
      }
    }
#endif
    __syncthreads();

    // ---- S = (Q/8) * K^T : 16x64 scores per wave, K-dim 64 = 2 chained WMMA ----
    v8f s[4];
#pragma unroll
    for (int nf = 0; nf < 4; ++nf) {
      v8f z = vzero;
      z = wmma_bf16(qfrag0, load_frag_b(sKt + nf * 16 * AT_LDK, AT_LDK, 0,  lane), z);
      z = wmma_bf16(qfrag1, load_frag_b(sKt + nf * 16 * AT_LDK, AT_LDK, 16, lane), z);
      s[nf] = z;
    }

    // ---- online softmax (row = half-wave of 16 lanes) ----
    float scl[8];
#pragma unroll
    for (int r = 0; r < 8; ++r) {
      float mx = fmaxf(fmaxf(s[0][r], s[1][r]), fmaxf(s[2][r], s[3][r]));
      mx = fmaxf(mx, __shfl_xor(mx, 1, 16));
      mx = fmaxf(mx, __shfl_xor(mx, 2, 16));
      mx = fmaxf(mx, __shfl_xor(mx, 4, 16));
      mx = fmaxf(mx, __shfl_xor(mx, 8, 16));
      const float nm = fmaxf(mrun[r], mx);
      scl[r] = __expf(mrun[r] - nm);
      mrun[r] = nm;
    }
#pragma unroll
    for (int nf = 0; nf < 4; ++nf)
#pragma unroll
      for (int r = 0; r < 8; ++r)
        s[nf][r] = __expf(s[nf][r] - mrun[r]);
#pragma unroll
    for (int r = 0; r < 8; ++r) {
      float sm = s[0][r] + s[1][r] + s[2][r] + s[3][r];
      sm += __shfl_xor(sm, 1, 16);
      sm += __shfl_xor(sm, 2, 16);
      sm += __shfl_xor(sm, 4, 16);
      sm += __shfl_xor(sm, 8, 16);
      lrun[r] = lrun[r] * scl[r] + sm;
    }
#pragma unroll
    for (int d = 0; d < 4; ++d)
#pragma unroll
      for (int r = 0; r < 8; ++r) oacc[d][r] *= scl[r];

    // ---- P (C-layout f32) -> LDS as bf16 A-tile [m][key] ----
#pragma unroll
    for (int nf = 0; nf < 4; ++nf)
#pragma unroll
      for (int r = 0; r < 8; ++r)
        pW[(r + hf * 8) * (2*AT_LDK) + nf * 16 + ln] = f2bf(s[nf][r]);
    __syncthreads();

    // ---- O += P * V ----
    v16bf p0 = load_frag_a(sP[wave], AT_LDK, 0,  lane);
    v16bf p1 = load_frag_a(sP[wave], AT_LDK, 16, lane);
#pragma unroll
    for (int df = 0; df < 4; ++df) {
      oacc[df] = wmma_bf16(p0, load_frag_b(sV + df * 16 * AT_LDK, AT_LDK, 0,  lane), oacc[df]);
      oacc[df] = wmma_bf16(p1, load_frag_b(sV + df * 16 * AT_LDK, AT_LDK, 16, lane), oacc[df]);
    }
    __syncthreads();
  }

  // ---- epilogue: O /= l, write y in [B,T,C] bf16 ----
#pragma unroll
  for (int df = 0; df < 4; ++df)
#pragma unroll
    for (int r = 0; r < 8; ++r) {
      const int t = qRow0 + r + hf * 8;
      const int d = df * 16 + ln;
      const float v = oacc[df][r] / lrun[r];
      Y[((size_t)b * Tdim + t) * Cdim + h * Ddim + d] = f2bf(v);
    }
}

// ----------------------------------------------------------------------------
extern "C" void kernel_launch(void* const* d_in, const int* in_sizes, int n_in,
                              void* d_out, int out_size, void* d_ws, size_t ws_size,
                              hipStream_t stream) {
  const float* x      = (const float*)d_in[0];
  const float* w_attn = (const float*)d_in[1];
  const float* b_attn = (const float*)d_in[2];
  const float* w_proj = (const float*)d_in[3];
  const float* b_proj = (const float*)d_in[4];
  float* out = (float*)d_out;

  char* ws = (char*)d_ws;
  size_t off = 0;
  auto walloc = [&](size_t bytes) -> unsigned short* {
    unsigned short* p = (unsigned short*)(ws + off);
    off += (bytes + 255) & ~(size_t)255;
    return p;
  };
  const size_t nBTC = (size_t)Bdim * Tdim * Cdim;              // 8.39M elements
  unsigned short* xb  = walloc(nBTC * 2);                      // x   bf16
  unsigned short* wab = walloc((size_t)Cdim * 3 * Cdim * 2);   // w_attn bf16
  unsigned short* wpb = walloc((size_t)Cdim * Cdim * 2);       // w_proj bf16
  unsigned short* qb  = walloc(nBTC * 2);                      // q [B,H,T,D] (pre-scaled)
  unsigned short* kb  = walloc(nBTC * 2);                      // k [B,H,T,D]
  unsigned short* vb  = walloc(nBTC * 2);                      // v [B,H,T,D]
  unsigned short* yb  = walloc(nBTC * 2);                      // attn out [B,T,C]

  cvt_f32_bf16<<<2048, 256, 0, stream>>>(x, xb, (int)nBTC);
  cvt_f32_bf16<<<2048, 256, 0, stream>>>(w_attn, wab, Cdim * 3 * Cdim);
  cvt_f32_bf16<<<1024, 256, 0, stream>>>(w_proj, wpb, Cdim * Cdim);

  dim3 gQKV(3 * Cdim / BN, (Bdim * Tdim) / BM);                 // (24, 64)
  gemm_bf16<0><<<gQKV, 256, 0, stream>>>(xb, wab, b_attn, qb, kb, vb, nullptr,
                                         3 * Cdim, Cdim);

  dim3 gA(Tdim / 64, Hdim, Bdim);                               // (32, 16, 4)
  attn_fa<<<gA, 128, 0, stream>>>(qb, kb, vb, yb);

  dim3 gP(Cdim / BN, (Bdim * Tdim) / BM);                       // (8, 64)
  gemm_bf16<1><<<gP, 256, 0, stream>>>(yb, wpb, b_proj, nullptr, nullptr, nullptr,
                                       out, Cdim, Cdim);
}
